// Correlation_87316685128078
// MI455X (gfx1250) — matrix operand order repaired and checked by reading, not compiled, not run on hardware
//
#include <hip/hip_runtime.h>

// FlowNetC correlation on gfx1250 via v_wmma_f32_16x16x32_f16.
// out[b, o, y, x] = (1/256) * sum_c in1[b,c,y,x] * in2_pad[b,c,y+dy,x+dx]
// B=8 C=256 H=96 W=128, dy,dx in {-20..20 step 2} -> 441 planes.
//
// Per block (b,y): in1 row -> LDS f16 [x][c] once, then A fragments held in
// registers across all 21 dy; per dy, in2 row y+dy -> double-buffered LDS f16
// [x'][c] (prefetched a phase ahead); 8 wave32s each compute a 16x64 f16-WMMA
// tile (M=16 x, N=64 shifted cols), software-pipelining B-fragment LDS loads
// one K-step ahead, then scatter the 21 valid even diagonals into an LDS
// staging buffer for coalesced global stores.

typedef _Float16 v8h  __attribute__((ext_vector_type(8)));
typedef _Float16 v16h __attribute__((ext_vector_type(16)));
typedef _Float16 h2   __attribute__((ext_vector_type(2)));
typedef float    v8f  __attribute__((ext_vector_type(8)));

#define CC   256
#define HH   96
#define WW   128
#define NDY  21
#define NDX  21
#define STRH 264   // f16 per LDS row (256 + pad; 528 B = 33*16B -> conflict-free b128)
#define BCOLS 176  // B cols: x' in [-20,147] -> 168, +8 pad for n-window overshoot
#define OSTR 132   // floats per staging row (128 + 4 pad)

static __device__ __forceinline__ v16h ld16(const _Float16* p0, const _Float16* p1)
{
    v8h lo = *(const v8h*)p0;
    v8h hi = *(const v8h*)p1;
    return __builtin_shufflevector(lo, hi, 0,1,2,3,4,5,6,7,8,9,10,11,12,13,14,15);
}

static __device__ __forceinline__ void
ldb4(const _Float16* bcol, int kk, v16h& b0, v16h& b1, v16h& b2, v16h& b3)
{
    const _Float16* bp = bcol + kk * 32;
    b0 = ld16(bp + 0 * 16 * STRH, bp + 0 * 16 * STRH + 8);
    b1 = ld16(bp + 1 * 16 * STRH, bp + 1 * 16 * STRH + 8);
    b2 = ld16(bp + 2 * 16 * STRH, bp + 2 * 16 * STRH + 8);
    b3 = ld16(bp + 3 * 16 * STRH, bp + 3 * 16 * STRH + 8);
}

static __device__ __forceinline__ void
prefetch_row(const float* __restrict__ in2, int b, int y2, int tid, size_t plane)
{
    // 256 c-rows x 512 B = 1024 cachelines of 128 B; 4 prefetches/thread.
    const float* base = in2 + (size_t)(b * CC) * plane + (size_t)y2 * WW;
#pragma unroll
    for (int i = 0; i < 4; ++i) {
        const int gid = i * 256 + tid;       // 0..1023
        const int c = gid >> 2, l = gid & 3;
        __builtin_prefetch(base + (size_t)c * plane + l * 32, 0, 0);
    }
}

static __device__ __forceinline__ void
load_row(const float* __restrict__ in2, int b, int y2, _Float16* buf, int tid, size_t plane)
{
    // gather in2[b, :, y2, :] -> buf[x+20][c]  (f32 -> f16, transpose)
    const int x     = tid & 127;
    const int chalf = (tid >> 7) << 1;       // 0 or 2
    const float* p2 = in2 + (size_t)(b * CC) * plane + (size_t)y2 * WW + x;
    _Float16* bcol  = buf + (x + 20) * STRH;
#pragma unroll 4
    for (int it = 0; it < 64; ++it) {
        const int c = it * 4 + chalf;
        const float f0 = p2[(size_t)c * plane];
        const float f1 = p2[(size_t)(c + 1) * plane];
        h2 p; p.x = (_Float16)f0; p.y = (_Float16)f1;
        *(h2*)(bcol + c) = p;
    }
}

__global__ __launch_bounds__(256) void
corr_wmma_kernel(const float* __restrict__ in1,
                 const float* __restrict__ in2,
                 float* __restrict__ out)
{
    extern __shared__ char smem_raw[];
    _Float16* As  = (_Float16*)smem_raw;          // [128][STRH]
    _Float16* Bs0 = As + 128 * STRH;              // [BCOLS][STRH] (dy even)
    _Float16* Bs1 = Bs0 + BCOLS * STRH;           // [BCOLS][STRH] (dy odd)
    float*    Os  = (float*)(Bs1 + BCOLS * STRH); // [NDX][OSTR]

    const int y   = blockIdx.x;   // 0..95
    const int b   = blockIdx.y;   // 0..7
    const int tid = threadIdx.x;  // 256 threads = 8 wave32

    const size_t plane = (size_t)HH * WW;

    if (y >= 20) prefetch_row(in2, b, y - 20, tid, plane);

    // zero both B buffers once (borders stay zero; interior rewritten per dy)
    {
        unsigned int* bz = (unsigned int*)Bs0;
        const int n = 2 * BCOLS * STRH / 2;       // dwords
        for (int i = tid; i < n; i += 256) bz[i] = 0u;
    }

    // in1 row y -> As (f32 -> f16, transpose to [x][c])
    {
        const int x     = tid & 127;
        const int chalf = (tid >> 7) << 1;
        const float* p1 = in1 + (size_t)(b * CC) * plane + (size_t)y * WW + x;
        _Float16* arow  = As + x * STRH;
#pragma unroll 4
        for (int it = 0; it < 64; ++it) {
            const int c = it * 4 + chalf;
            const float f0 = p1[(size_t)c * plane];
            const float f1 = p1[(size_t)(c + 1) * plane];
            h2 p; p.x = (_Float16)f0; p.y = (_Float16)f1;
            *(h2*)(arow + c) = p;
        }
    }
    if (y >= 20) load_row(in2, b, y - 20, Bs0, tid, plane);  // preload dyi=0
    __syncthreads();

    const int wave = tid >> 5;
    const int lane = tid & 31;
    const int x0   = wave << 4;                  // 16-pixel M tile per wave
    const int lm   = lane & 15;
    const int hs   = lane >> 4;                  // half-select (0/1)

    // A fragments are dy-invariant: hold all 8 K-steps in registers (64 VGPRs).
    // A 16x32 layout: K = k0 + hs*16 + {0..15} (contiguous, matches B).
    v16h aK[8];
    {
        const _Float16* arow = As + (x0 + lm) * STRH + hs * 16;
#pragma unroll
        for (int kk = 0; kk < 8; ++kk) {
            const _Float16* ap = arow + kk * 32;
            aK[kk] = ld16(ap, ap + 8);
        }
    }

    for (int dyi = 0; dyi < NDY; ++dyi) {
        _Float16* cur = (dyi & 1) ? Bs1 : Bs0;
        _Float16* nxt = (dyi & 1) ? Bs0 : Bs1;
        const int  y2c    = y + dyi * 2 - 20;
        const int  y2n    = y2c + 2;
        const bool validc = (y2c >= 0) && (y2c < HH);
        const bool validn = (dyi < NDY - 1) && (y2n >= 0) && (y2n < HH);

        if (validn) prefetch_row(in2, b, y2n, tid, plane);

        if (validc) {
            v8f acc0 = (v8f){0.f,0.f,0.f,0.f,0.f,0.f,0.f,0.f};
            v8f acc1 = acc0, acc2 = acc0, acc3 = acc0;

            // B col for tile t: x' index (x0 + t*16 + lm); K contiguous per half
            const _Float16* bcol = cur + (x0 + lm) * STRH + hs * 16;

            // copy-free 2-deep pipeline: loads of K-step k+1 precede WMMAs of k
            v16h p0, p1, p2, p3, q0, q1, q2, q3;
            ldb4(bcol, 0, p0, p1, p2, p3);
#pragma unroll
            for (int kk = 0; kk < 8; kk += 2) {
                ldb4(bcol, kk + 1, q0, q1, q2, q3);
                acc0 = __builtin_amdgcn_wmma_f32_16x16x32_f16(
                           false, aK[kk], false, p0, (short)0, acc0, false, false);
                acc1 = __builtin_amdgcn_wmma_f32_16x16x32_f16(
                           false, aK[kk], false, p1, (short)0, acc1, false, false);
                acc2 = __builtin_amdgcn_wmma_f32_16x16x32_f16(
                           false, aK[kk], false, p2, (short)0, acc2, false, false);
                acc3 = __builtin_amdgcn_wmma_f32_16x16x32_f16(
                           false, aK[kk], false, p3, (short)0, acc3, false, false);
                if (kk + 2 < 8) ldb4(bcol, kk + 2, p0, p1, p2, p3);
                acc0 = __builtin_amdgcn_wmma_f32_16x16x32_f16(
                           false, aK[kk + 1], false, q0, (short)0, acc0, false, false);
                acc1 = __builtin_amdgcn_wmma_f32_16x16x32_f16(
                           false, aK[kk + 1], false, q1, (short)0, acc1, false, false);
                acc2 = __builtin_amdgcn_wmma_f32_16x16x32_f16(
                           false, aK[kk + 1], false, q2, (short)0, acc2, false, false);
                acc3 = __builtin_amdgcn_wmma_f32_16x16x32_f16(
                           false, aK[kk + 1], false, q3, (short)0, acc3, false, false);
            }

            // scatter valid diagonals (even dx, |dx|<=20) into LDS staging
            const float scale = 1.0f / 256.0f;
            v8f accs[4] = {acc0, acc1, acc2, acc3};
#pragma unroll
            for (int t = 0; t < 4; ++t) {
#pragma unroll
                for (int r = 0; r < 8; ++r) {
                    const int M  = r + hs * 8;          // D: VGPR r, lane half -> M
                    const int dx = t * 16 + lm - 20 - M;
                    if (!(dx & 1) && dx >= -20 && dx <= 20) {
                        Os[((dx + 20) >> 1) * OSTR + x0 + M] = accs[t][r] * scale;
                    }
                }
            }
        }
        __syncthreads();

        // gather next row (prefetched -> cache-hot) while writing out results
        if (validn) load_row(in2, b, y2n, nxt, tid, plane);

        // coalesced writeback: 21 planes x 128 contiguous floats
        {
            const int obase = ((b * (NDY * NDX) + dyi * NDX) * HH + y) * WW;
            for (int i = tid; i < NDX * WW; i += 256) {
                const int j = i >> 7;
                const int x = i & 127;
                out[obase + j * HH * WW + x] = validc ? Os[j * OSTR + x] : 0.0f;
            }
        }
        __syncthreads();
    }
}

extern "C" void kernel_launch(void* const* d_in, const int* in_sizes, int n_in,
                              void* d_out, int out_size, void* d_ws, size_t ws_size,
                              hipStream_t stream)
{
    (void)in_sizes; (void)n_in; (void)d_ws; (void)ws_size; (void)out_size;
    const float* in1 = (const float*)d_in[0];
    const float* in2 = (const float*)d_in[1];
    float* out = (float*)d_out;

    const size_t shmem = (size_t)(128 * STRH + 2 * BCOLS * STRH) * sizeof(_Float16)
                       + (size_t)(NDX * OSTR) * sizeof(float); // 264,528 B < 320 KB

    dim3 grid(HH, 8);   // (y, b)
    dim3 block(256);    // 8 wave32
    corr_wmma_kernel<<<grid, block, shmem, stream>>>(in1, in2, out);
}
